// AttnDecoderRNN_70231305224497
// MI455X (gfx1250) — compile-verified, matrix operand order checked by reading, not run on hardware
//
#include <hip/hip_runtime.h>

#define HD   256          // hidden
#define OUTW 3
#define NPRED 64
#define BB   256          // batch
#define SS   512          // seq len
#define INW  (HD + OUTW)  // 259
#define KC   512          // combined K for gates GEMM (ctx | h)
#define G4   (4 * HD)     // 1024
#define CH   64           // UK rows per TDM chunk (32 KB bf16)

typedef __attribute__((ext_vector_type(16))) __bf16 v16bf;
typedef __attribute__((ext_vector_type(8)))  __bf16 v8bf;
typedef __attribute__((ext_vector_type(8)))  float  v8f;
typedef __attribute__((ext_vector_type(4)))  unsigned int v4u;
typedef __attribute__((ext_vector_type(8)))  int v8i;
typedef __attribute__((ext_vector_type(4)))  int v4i;

#if defined(__has_builtin)
#if __has_builtin(__builtin_amdgcn_tensor_load_to_lds) && \
    __has_builtin(__builtin_amdgcn_s_wait_tensorcnt)
#define HAVE_TDM 1
#endif
#endif

// ---------------------------------------------------------------------------
// WMMA helpers (CDNA5 wave32, V_WMMA_F32_16X16X32_BF16)
// A 16x32 bf16: lane L -> row m=L&15, half=L>>4; VGPR v holds K pair
//   base(v) = (v<4 ? 2v : 16+2(v-4)) + half*8
// B 32x16 bf16: mirrored (N in lanes), same K mapping.
// C/D 16x16 f32: lane L -> col n=L&15; VGPR r -> row m = r + (L>>4)*8
// ---------------------------------------------------------------------------
__device__ __forceinline__ v16bf load_tile16(const __bf16* __restrict__ p, int ld) {
  const int lane = threadIdx.x & 31;
  const int m = lane & 15, hf = lane >> 4;
  const __bf16* row = p + (size_t)m * ld + hf * 8;
  v16bf t;
#pragma unroll
  for (int v = 0; v < 8; ++v) {
    const int k = (v < 4) ? (2 * v) : (16 + 2 * (v - 4));
    t[2 * v]     = row[k];
    t[2 * v + 1] = row[k + 1];
  }
  return t;
}

__device__ __forceinline__ v8f wmma_bf16(v16bf a, v16bf b, v8f c) {
  return __builtin_amdgcn_wmma_f32_16x16x32_bf16(false, a, false, b, (short)0, c,
                                                 false, false);
}

__device__ __forceinline__ float sigf(float x) { return 1.f / (1.f + __expf(-x)); }

#ifdef HAVE_TDM
// 2D tensor-DMA: load tile (tile0 x tile1 elements, 2-byte) from global row-major
// tensor (leading-dim stride0 elements) into LDS at byte offset lds_off.
__device__ __forceinline__ void tdm_load_2d(unsigned lds_off, const void* gptr,
                                            unsigned td0, unsigned td1,
                                            unsigned tile0, unsigned tile1,
                                            unsigned st0) {
  const unsigned long long ga = (unsigned long long)gptr;
  // D# group 0: count=1 | lds_addr | global_addr[56:0] | type=2
  v4u g0 = { 1u, lds_off, (unsigned)(ga & 0xFFFFFFFFu),
             (unsigned)((ga >> 32) & 0x1FFFFFFu) | 0x80000000u };
  // D# group 1: data_size=1(2B); tensor_dim0/1; tile_dim0/1; dim0 stride
  v8i g1 = { (int)(1u << 16),
             (int)((td0 & 0xFFFFu) << 16),
             (int)(((td0 >> 16) & 0xFFFFu) | ((td1 & 0xFFFFu) << 16)),
             (int)(((td1 >> 16) & 0xFFFFu) | ((tile0 & 0xFFFFu) << 16)),
             (int)(tile1 & 0xFFFFu),
             (int)st0, 0, 0 };
  v4i z = { 0, 0, 0, 0 };
#if __clang_major__ >= 23
  v8i z8 = { 0, 0, 0, 0, 0, 0, 0, 0 };
  __builtin_amdgcn_tensor_load_to_lds(g0, g1, z, z, z8, 0);
#else
  __builtin_amdgcn_tensor_load_to_lds(g0, g1, z, z, 0);
#endif
}
#endif

// ---------------------------------------------------------------------------
// One-time prep kernels
// ---------------------------------------------------------------------------
__global__ void k_f32_to_bf16(const float* __restrict__ src, __bf16* __restrict__ dst,
                              int n) {
  int i = (blockIdx.x * blockDim.x + threadIdx.x) * 4;
  if (i + 3 < n) {
#pragma unroll
    for (int j = 0; j < 4; ++j) dst[i + j] = (__bf16)src[i + j];
  } else {
    for (int j = 0; j < 4 && i + j < n; ++j) dst[i + j] = (__bf16)src[i + j];
  }
}

// Wcomb[n][k] = k<HD ? W_ih[n][3+k] : W_hh[n][k-HD]   (bf16, row n over 4H)
__global__ void k_pack_wcomb(const float* __restrict__ Wih,
                             const float* __restrict__ Whh,
                             __bf16* __restrict__ Wc) {
  int i = blockIdx.x * blockDim.x + threadIdx.x;
  if (i >= G4 * KC) return;
  int n = i / KC, k = i % KC;
  float v = (k < HD) ? Wih[(size_t)n * INW + OUTW + k]
                     : Whh[(size_t)n * HD + (k - HD)];
  Wc[i] = (__bf16)v;
}

__global__ void k_init_state(const float* __restrict__ h0, const float* __restrict__ c0,
                             __bf16* __restrict__ xcomb, float* __restrict__ hf,
                             float* __restrict__ cf, float* __restrict__ xin) {
  int i = blockIdx.x * blockDim.x + threadIdx.x;
  if (i < BB * HD) {
    int b = i / HD, hh = i % HD;
    float v = h0[i];
    xcomb[(size_t)b * KC + HD + hh] = (__bf16)v;  // h lives in columns [HD,2HD)
    hf[i] = v;
    cf[i] = c0[i];
  }
  if (i < BB * OUTW) xin[i] = 0.0f;  // SOS token = 0
}

// UK[b*s, g] = sum_h enc_bf16[b*s, h] * Ua[g, h] + bua[g]  -> bf16
// M = B*S = 131072, N = 256, K = 256. One 16x16 tile per wave.
__global__ void k_uk_gemm(const __bf16* __restrict__ A, const __bf16* __restrict__ Ub,
                          const float* __restrict__ bua, __bf16* __restrict__ UK) {
  const int wave = (blockIdx.x * blockDim.x + threadIdx.x) >> 5;
  const int NT = HD / 16;  // 16
  const int tm = wave / NT, tn = wave % NT;
  const __bf16* arow = A + (size_t)tm * 16 * HD;
  const __bf16* brow = Ub + (size_t)tn * 16 * HD;
  v8f acc = {};
#pragma unroll
  for (int k0 = 0; k0 < HD; k0 += 32)
    acc = wmma_bf16(load_tile16(arow + k0, HD), load_tile16(brow + k0, HD), acc);
  const int lane = threadIdx.x & 31;
  const int n = lane & 15, hf = lane >> 4;
  const int col = tn * 16 + n;
  const float bias = bua[col];
#pragma unroll
  for (int r = 0; r < 8; ++r) {
    size_t row = (size_t)tm * 16 + hf * 8 + r;
    UK[row * HD + col] = (__bf16)(acc[r] + bias);
  }
}

// ---------------------------------------------------------------------------
// Per-step kernels
// ---------------------------------------------------------------------------
// wq[b,g] = h[b,:]·Wa[g,:] + ba[g]   (M=256,N=256,K=256)
__global__ void k_wq_gemm(const __bf16* __restrict__ xcomb,
                          const __bf16* __restrict__ Wab,
                          const float* __restrict__ ba, float* __restrict__ wq) {
  const int wave = (blockIdx.x * blockDim.x + threadIdx.x) >> 5;
  const int NT = HD / 16;
  const int tm = wave / NT, tn = wave % NT;
  const __bf16* arow = xcomb + (size_t)tm * 16 * KC + HD;  // h section
  const __bf16* brow = Wab + (size_t)tn * 16 * HD;
  v8f acc = {};
#pragma unroll
  for (int k0 = 0; k0 < HD; k0 += 32)
    acc = wmma_bf16(load_tile16(arow + k0, KC), load_tile16(brow + k0, HD), acc);
  const int lane = threadIdx.x & 31;
  const int n = lane & 15, hf = lane >> 4;
  const int col = tn * 16 + n;
  const float bias = ba[col];
#pragma unroll
  for (int r = 0; r < 8; ++r) {
    int row = tm * 16 + hf * 8 + r;
    wq[(size_t)row * HD + col] = acc[r] + bias;
  }
}

// Fused attention: scores = UK[b]·wq[b], softmax over S, ctx = w·enc[b].
// One 512-thread (16 wave32) block per batch row. UK chunks streamed into LDS
// by the Tensor Data Mover (double buffered), ctx reads 16B/lane from L2.
__global__ void __launch_bounds__(512) k_attn(
    const __bf16* __restrict__ UK, const __bf16* __restrict__ encb,
    const float* __restrict__ wq, __bf16* __restrict__ xcomb,
    float* __restrict__ attn_out, int step) {
  __shared__ float wq_s[HD];
  __shared__ float sc[SS];
  __shared__ float red[16];
  __shared__ float cpart[16][HD];
  const int b = blockIdx.x;
  const int t = threadIdx.x;
  const int lane = t & 31, wv = t >> 5;

  if (t < HD) wq_s[t] = wq[(size_t)b * HD + t];
  __syncthreads();

  // each lane owns 8 contiguous k; hoist matching wq into registers
  float wr[8];
#pragma unroll
  for (int j = 0; j < 8; ++j) wr[j] = wq_s[lane * 8 + j];

  const __bf16* ukb = UK + (size_t)b * SS * HD;

#ifdef HAVE_TDM
  // ---- scores via TDM: stream 64-row (32KB) UK chunks into LDS, dbl-buffered
  __shared__ __attribute__((aligned(16))) __bf16 ukbuf[2][CH * HD];
  const unsigned lds_b[2] = {
      (unsigned)(unsigned long long)(const void*)&ukbuf[0][0],
      (unsigned)(unsigned long long)(const void*)&ukbuf[1][0]};
  if (wv == 0) {
    tdm_load_2d(lds_b[0], ukb, HD, SS, HD, CH, HD);
    __builtin_amdgcn_s_wait_tensorcnt(0);
  }
  __syncthreads();
  const int NCHUNK = SS / CH;  // 8
  for (int c = 0; c < NCHUNK; ++c) {
    if (wv == 0 && c + 1 < NCHUNK)
      tdm_load_2d(lds_b[(c + 1) & 1], ukb + (size_t)(c + 1) * CH * HD,
                  HD, SS, HD, CH, HD);
    const __bf16* bufc = ukbuf[c & 1];
#pragma unroll
    for (int q = 0; q < CH / 16; ++q) {  // 4 rows per wave per chunk
      const int r = wv * (CH / 16) + q;
      v8bf ev = *(const v8bf*)(bufc + r * HD + lane * 8);  // ds_load_b128
      float acc = 0.f;
#pragma unroll
      for (int j = 0; j < 8; ++j) acc += wr[j] * (float)ev[j];
#pragma unroll
      for (int off = 16; off; off >>= 1) acc += __shfl_xor(acc, off);
      if (lane == 0) sc[c * CH + r] = acc;
    }
    if (wv == 0 && c + 1 < NCHUNK) __builtin_amdgcn_s_wait_tensorcnt(0);
    __syncthreads();
  }
#else
  // ---- scores fallback: 16B/lane coalesced global loads
  for (int j = 0; j < 32; ++j) {
    const int s = wv * 32 + j;
    const __bf16* row = ukb + (size_t)s * HD;
    if (j + 1 < 32) __builtin_prefetch(row + HD, 0, 1);
    v8bf ev = *(const v8bf*)(row + lane * 8);  // global_load_b128
    float acc = 0.f;
#pragma unroll
    for (int k = 0; k < 8; ++k) acc += wr[k] * (float)ev[k];
#pragma unroll
    for (int off = 16; off; off >>= 1) acc += __shfl_xor(acc, off);
    if (lane == 0) sc[s] = acc;
  }
  __syncthreads();
#endif

  // ---- softmax over S=512 (wave shuffle + 16-wide LDS tree)
  const float v = sc[t];
  float m = v;
#pragma unroll
  for (int off = 16; off; off >>= 1) m = fmaxf(m, __shfl_xor(m, off));
  if (lane == 0) red[wv] = m;
  __syncthreads();
  if (t < 32) {
    float x = (t < 16) ? red[t] : -3.402823466e38f;
#pragma unroll
    for (int off = 8; off; off >>= 1) x = fmaxf(x, __shfl_xor(x, off));
    if (t == 0) red[0] = x;
  }
  __syncthreads();
  m = red[0];
  __syncthreads();
  const float e = __expf(v - m);
  float ssum = e;
#pragma unroll
  for (int off = 16; off; off >>= 1) ssum += __shfl_xor(ssum, off);
  if (lane == 0) red[wv] = ssum;
  __syncthreads();
  if (t < 32) {
    float x = (t < 16) ? red[t] : 0.f;
#pragma unroll
    for (int off = 8; off; off >>= 1) x += __shfl_xor(x, off);
    if (t == 0) red[0] = x;
  }
  __syncthreads();
  const float wgt = e / red[0];
  sc[t] = wgt;
  attn_out[((size_t)b * NPRED + step) * SS + t] = wgt;
  __syncthreads();

  // ---- ctx[b,h] = sum_s w[s]*enc[b,s,h]; 16 s-rows in flight, 16B/lane
  const __bf16* eb = encb + (size_t)b * SS * HD;
  const int sg = t >> 5;             // s-group 0..15
  const int hbase = (t & 31) * 8;    // 8 contiguous h per lane
  float acc8[8];
#pragma unroll
  for (int j = 0; j < 8; ++j) acc8[j] = 0.f;
  for (int s = sg; s < SS; s += 16) {
    if (((s >> 4) & 7) == 0)
      __builtin_prefetch(eb + (size_t)(s + 128) * HD + hbase, 0, 1);
    v8bf ev = *(const v8bf*)(eb + (size_t)s * HD + hbase);  // global_load_b128
    const float w = sc[s];
#pragma unroll
    for (int j = 0; j < 8; ++j) acc8[j] += w * (float)ev[j];
  }
#pragma unroll
  for (int j = 0; j < 8; ++j) cpart[sg][hbase + j] = acc8[j];
  __syncthreads();
  if (t < HD) {
    float cx = 0.f;
#pragma unroll
    for (int g = 0; g < 16; ++g) cx += cpart[g][t];
    xcomb[(size_t)b * KC + t] = (__bf16)cx;  // ctx in cols [0,HD)
  }
}

// gates[b,n] = [ctx|h]·Wcomb[n,:] + b_ih[n]+b_hh[n] + x_in[b,:3]·W_ih[n,:3]
// M=256, N=1024, K=512
__global__ void k_gates(const __bf16* __restrict__ xcomb, const __bf16* __restrict__ Wc,
                        const float* __restrict__ b_ih, const float* __restrict__ b_hh,
                        const float* __restrict__ Wih, const float* __restrict__ xin,
                        float* __restrict__ gates) {
  const int wave = (blockIdx.x * blockDim.x + threadIdx.x) >> 5;
  const int NT = G4 / 16;  // 64
  const int tm = wave / NT, tn = wave % NT;
  const __bf16* arow = xcomb + (size_t)tm * 16 * KC;
  const __bf16* brow = Wc + (size_t)tn * 16 * KC;
  v8f acc = {};
  for (int k0 = 0; k0 < KC; k0 += 32)
    acc = wmma_bf16(load_tile16(arow + k0, KC), load_tile16(brow + k0, KC), acc);
  const int lane = threadIdx.x & 31;
  const int n = lane & 15, hf = lane >> 4;
  const int col = tn * 16 + n;
  const float bias = b_ih[col] + b_hh[col];
  const float w0 = Wih[(size_t)col * INW + 0];
  const float w1 = Wih[(size_t)col * INW + 1];
  const float w2 = Wih[(size_t)col * INW + 2];
#pragma unroll
  for (int r = 0; r < 8; ++r) {
    const int row = tm * 16 + hf * 8 + r;
    gates[(size_t)row * G4 + col] =
        acc[r] + bias + xin[row * 3] * w0 + xin[row * 3 + 1] * w1 +
        xin[row * 3 + 2] * w2;
  }
}

__global__ void k_lstm(const float* __restrict__ gates, float* __restrict__ cf,
                       float* __restrict__ hf, __bf16* __restrict__ xcomb) {
  int i = blockIdx.x * blockDim.x + threadIdx.x;
  if (i >= BB * HD) return;
  int b = i / HD, hh = i % HD;
  const float* g = gates + (size_t)b * G4;
  float ig = sigf(g[hh]);
  float fg = sigf(g[HD + hh]);
  float gg = tanhf(g[2 * HD + hh]);
  float og = sigf(g[3 * HD + hh]);
  float c = fg * cf[i] + ig * gg;
  float h = og * tanhf(c);
  cf[i] = c;
  hf[i] = h;
  xcomb[(size_t)b * KC + HD + hh] = (__bf16)h;  // feed next step's wq/gates
}

// pred[b,o] = h[b,:]·Wo[o,:] + bo[o]; also writes x_in for next step
__global__ void k_pred(const float* __restrict__ hf, const float* __restrict__ Wo,
                       const float* __restrict__ bo, float* __restrict__ pred_out,
                       float* __restrict__ xin, int step) {
  __shared__ float sdata[256];
  const int b = blockIdx.x, t = threadIdx.x;
  const float h = hf[(size_t)b * HD + t];
  for (int o = 0; o < OUTW; ++o) {
    sdata[t] = h * Wo[(size_t)o * HD + t];
    __syncthreads();
    for (int s = 128; s; s >>= 1) {
      if (t < s) sdata[t] += sdata[t + s];
      __syncthreads();
    }
    if (t == 0) {
      float p = sdata[0] + bo[o];
      pred_out[((size_t)b * NPRED + step) * OUTW + o] = p;
      xin[b * OUTW + o] = p;
    }
    __syncthreads();
  }
}

// ---------------------------------------------------------------------------
extern "C" void kernel_launch(void* const* d_in, const int* in_sizes, int n_in,
                              void* d_out, int out_size, void* d_ws, size_t ws_size,
                              hipStream_t stream) {
  (void)in_sizes; (void)n_in; (void)out_size; (void)ws_size;
  const float* enc  = (const float*)d_in[0];
  const float* h0   = (const float*)d_in[1];
  const float* c0   = (const float*)d_in[2];
  const float* Wa   = (const float*)d_in[3];
  const float* ba   = (const float*)d_in[4];
  const float* Ua   = (const float*)d_in[5];
  const float* bua  = (const float*)d_in[6];
  const float* W_ih = (const float*)d_in[7];
  const float* W_hh = (const float*)d_in[8];
  const float* b_ih = (const float*)d_in[9];
  const float* b_hh = (const float*)d_in[10];
  const float* Wo   = (const float*)d_in[11];
  const float* bo   = (const float*)d_in[12];

  float* pred_out = (float*)d_out;                          // [B, N, OUT]
  float* hid_out  = pred_out + (size_t)BB * NPRED * OUTW;   // [1, B, H]
  float* attn_out = hid_out + (size_t)BB * HD;              // [B, N, S]

  char* ws = (char*)d_ws;
  size_t off = 0;
  auto alloc = [&](size_t bytes) {
    char* p = ws + off;
    off += (bytes + 255) & ~(size_t)255;
    return p;
  };
  __bf16* encb  = (__bf16*)alloc((size_t)BB * SS * HD * 2);  // 67 MB (L2-resident)
  __bf16* UK    = (__bf16*)alloc((size_t)BB * SS * HD * 2);  // 67 MB (L2-resident)
  __bf16* Wab   = (__bf16*)alloc((size_t)HD * HD * 2);
  __bf16* Uab   = (__bf16*)alloc((size_t)HD * HD * 2);
  __bf16* Wc    = (__bf16*)alloc((size_t)G4 * KC * 2);
  __bf16* xcomb = (__bf16*)alloc((size_t)BB * KC * 2);
  float*  wq    = (float*)alloc((size_t)BB * HD * 4);
  float*  gates = (float*)alloc((size_t)BB * G4 * 4);
  float*  cf    = (float*)alloc((size_t)BB * HD * 4);
  float*  hf    = (float*)alloc((size_t)BB * HD * 4);
  float*  xin   = (float*)alloc((size_t)BB * OUTW * 4);

  // ---- one-time prep
  { int n = BB * SS * HD; k_f32_to_bf16<<<(n / 4 + 255) / 256, 256, 0, stream>>>(enc, encb, n); }
  { int n = HD * HD;      k_f32_to_bf16<<<(n / 4 + 255) / 256, 256, 0, stream>>>(Wa, Wab, n); }
  { int n = HD * HD;      k_f32_to_bf16<<<(n / 4 + 255) / 256, 256, 0, stream>>>(Ua, Uab, n); }
  { int n = G4 * KC;      k_pack_wcomb<<<(n + 255) / 256, 256, 0, stream>>>(W_ih, W_hh, Wc); }
  k_init_state<<<(BB * HD + 255) / 256, 256, 0, stream>>>(h0, c0, xcomb, hf, cf, xin);
  // UK = enc @ Ua^T + bua : 131072x256x256 bf16 WMMA GEMM
  k_uk_gemm<<<(BB * SS / 16) * (HD / 16) / 8, 256, 0, stream>>>(encb, Uab, bua, UK);

  // ---- 64 autoregressive steps
  for (int step = 0; step < NPRED; ++step) {
    k_wq_gemm<<<(BB / 16) * (HD / 16) / 8, 256, 0, stream>>>(xcomb, Wab, ba, wq);
    k_attn<<<BB, 512, 0, stream>>>(UK, encb, wq, xcomb, attn_out, step);
    k_gates<<<(BB / 16) * (G4 / 16) / 8, 256, 0, stream>>>(xcomb, Wc, b_ih, b_hh,
                                                           W_ih, xin, gates);
    k_lstm<<<(BB * HD + 255) / 256, 256, 0, stream>>>(gates, cf, hf, xcomb);
    k_pred<<<BB, 256, 0, stream>>>(hf, Wo, bo, pred_out, xin, step);
  }

  hipMemcpyAsync(hid_out, hf, (size_t)BB * HD * 4, hipMemcpyDeviceToDevice, stream);
}
